// IntrospectiveFlashDiff2_60601988547137
// MI455X (gfx1250) — matrix operand order repaired and checked by reading, not compiled, not run on hardware
//
#include <hip/hip_runtime.h>

typedef __bf16 bf16_t;
typedef __attribute__((ext_vector_type(16))) __bf16 v16bf;
typedef __attribute__((ext_vector_type(8)))  __bf16 v8bf;
typedef __attribute__((ext_vector_type(8)))  float  v8f;

#define BQ 2
#define TQ 1024
#define DQ 1024
#define HQ 16
#define HKV 8
#define HD 64
#define MAXKV 3072
#define EPSQ 1e-5f

// ---------- fragment helpers (CDNA5 WMMA 16x16x32 bf16 layouts) ----------

__device__ __forceinline__ v16bf ld_a_frag(const bf16_t* p) {
  // lane<16: K {0..7,16..23} ; lane>=16 (caller adds +8): K {8..15,24..31}
  v8bf lo = *(const v8bf*)(p);
  v8bf hi = *(const v8bf*)(p + 16);
  return __builtin_shufflevector(lo, hi, 0,1,2,3,4,5,6,7,8,9,10,11,12,13,14,15);
}
__device__ __forceinline__ v16bf ld_b_frag(const bf16_t* p) {
  // lane<16: K {0..7,16..23} ; lane>=16 (caller adds +8): K {8..15,24..31}
  v8bf lo = *(const v8bf*)(p);
  v8bf hi = *(const v8bf*)(p + 16);
  return __builtin_shufflevector(lo, hi, 0,1,2,3,4,5,6,7,8,9,10,11,12,13,14,15);
}
__device__ __forceinline__ v8f wmma_bf16(v16bf a, v16bf b, v8f c) {
  return __builtin_amdgcn_wmma_f32_16x16x32_bf16(false, a, false, b, (short)0, c,
                                                 false, false);
}

// ---------- elementwise utility kernels ----------

__global__ void f32_to_bf16_kernel(const float* __restrict__ s,
                                   bf16_t* __restrict__ d, int n) {
  int i = blockIdx.x * blockDim.x + threadIdx.x;
  if (i < n) d[i] = (bf16_t)s[i];
}

__global__ void zero_f32_kernel(float* __restrict__ p, int n) {
  int i = blockIdx.x * blockDim.x + threadIdx.x;
  if (i < n) p[i] = 0.0f;
}

// ---------- register-blocked WMMA GEMM: C[M,N] = A[M,K] @ W[N,K]^T ----------
// one wave computes a 64x32 tile (4 M-subtiles x 2 N-subtiles, 8 WMMA / k-step)
// output row m -> (b = m / Tchunk, t = m % Tchunk) -> row (b*obs + roff + t)

__global__ void __launch_bounds__(256)
wmma_gemm_bf16(const bf16_t* __restrict__ A, const bf16_t* __restrict__ W,
               float* __restrict__ C, int M, int N, int K,
               int Tchunk, int obs, int roff) {
  int tiles_n = N >> 5;
  int tiles_m = M >> 6;
  int tile = blockIdx.x * 8 + (threadIdx.x >> 5);
  if (tile >= tiles_m * tiles_n) return;
  int lane = threadIdx.x & 31;
  int hf = lane >> 4, l16 = lane & 15;
  int mi = tile / tiles_n, ni = tile - mi * tiles_n;
  int m0 = mi << 6, n0 = ni << 5;

  const bf16_t* arow = A + (size_t)(m0 + l16) * K + hf * 8;
  const bf16_t* wrow = W + (size_t)(n0 + l16) * K + hf * 8;
  v8f c[8];
#pragma unroll
  for (int i = 0; i < 8; ++i) c[i] = (v8f){};

  for (int k0 = 0; k0 < K; k0 += 32) {
    __builtin_prefetch(arow + k0 + 128, 0, 1);
    __builtin_prefetch(wrow + k0 + 128, 0, 1);
    v16bf a0 = ld_a_frag(arow + k0);
    v16bf a1 = ld_a_frag(arow + (size_t)16 * K + k0);
    v16bf a2 = ld_a_frag(arow + (size_t)32 * K + k0);
    v16bf a3 = ld_a_frag(arow + (size_t)48 * K + k0);
    v16bf b0 = ld_b_frag(wrow + k0);
    v16bf b1 = ld_b_frag(wrow + (size_t)16 * K + k0);
    c[0] = wmma_bf16(a0, b0, c[0]);
    c[1] = wmma_bf16(a0, b1, c[1]);
    c[2] = wmma_bf16(a1, b0, c[2]);
    c[3] = wmma_bf16(a1, b1, c[3]);
    c[4] = wmma_bf16(a2, b0, c[4]);
    c[5] = wmma_bf16(a2, b1, c[5]);
    c[6] = wmma_bf16(a3, b0, c[6]);
    c[7] = wmma_bf16(a3, b1, c[7]);
  }
#pragma unroll
  for (int im = 0; im < 4; ++im) {
#pragma unroll
    for (int in = 0; in < 2; ++in) {
      v8f cc = c[im * 2 + in];
      int mb = m0 + im * 16 + hf * 8;
#pragma unroll
      for (int r = 0; r < 8; ++r) {
        int m = mb + r;
        int bb = m / Tchunk, t = m - bb * Tchunk;
        C[((size_t)bb * obs + roff + t) * N + n0 + in * 16 + l16] = cc[r];
      }
    }
  }
}

// ---------- rope on Q (f32 -> bf16), positions 0..T-1 ----------

__global__ void rope_q_kernel(const float* __restrict__ Q,
                              const float* __restrict__ cosT,
                              const float* __restrict__ sinT,
                              bf16_t* __restrict__ Qr, int n) {
  int idx = blockIdx.x * blockDim.x + threadIdx.x;
  if (idx >= n) return;
  int j = idx & 31;
  int h = (idx >> 5) & 15;
  int t = (idx >> 9) & 1023;
  int b = idx >> 19;
  size_t base = (((size_t)(b * TQ + t)) * HQ + h) * HD;
  float x1 = Q[base + j], x2 = Q[base + 32 + j];
  float cc = cosT[t * 32 + j], ss = sinT[t * 32 + j];
  Qr[base + j]      = (bf16_t)(x1 * cc - x2 * ss);
  Qr[base + 32 + j] = (bf16_t)(x2 * cc + x1 * ss);
}

// ---------- rope on whole K cache (f32 -> bf16), positions 0..L-1 ----------

__global__ void rope_k_kernel(const float* __restrict__ Kc,
                              const float* __restrict__ cosT,
                              const float* __restrict__ sinT,
                              bf16_t* __restrict__ Kr, int L, int n) {
  int idx = blockIdx.x * blockDim.x + threadIdx.x;
  if (idx >= n) return;
  int j = idx & 31;
  int hkv = (idx >> 5) & 7;
  int rest = idx >> 8;
  int l = rest % L;
  int b = rest / L;
  size_t base = (((size_t)(b * MAXKV + l)) * HKV + hkv) * HD;
  float x1 = Kc[base + j], x2 = Kc[base + 32 + j];
  float cc = cosT[l * 32 + j], ss = sinT[l * 32 + j];
  Kr[base + j]      = (bf16_t)(x1 * cc - x2 * ss);
  Kr[base + 32 + j] = (bf16_t)(x2 * cc + x1 * ss);
}

// ---------- transpose layer-V (f32, (B,T,HKV,HD)) into Vt bf16 (B,HKV,HD,MAXKV)

__global__ void v_transpose_kernel(const float* __restrict__ V,
                                   bf16_t* __restrict__ Vt, int koff, int n) {
  int idx = blockIdx.x * blockDim.x + threadIdx.x;
  if (idx >= n) return;
  int c = idx & 63;
  int hkv = (idx >> 6) & 7;
  int t = (idx >> 9) & 1023;
  int b = idx >> 19;
  float v = V[(((size_t)(b * TQ + t)) * HKV + hkv) * HD + c];
  Vt[(((size_t)(b * HKV + hkv)) * HD + c) * MAXKV + koff + t] = (bf16_t)v;
}

// ---------- flash attention ----------
// Block = 8 waves on 8 consecutive q-tiles of the SAME (b,h): the block shares
// each 32-key K/V chunk, cooperatively staged into LDS via async global->LDS
// DMA (ASYNCcnt), double-buffered so the next chunk's DMA overlaps WMMA work.

#define KPAD 72   // LDS row stride for K chunk (32 rows x 64 d)
#define VPAD 40   // LDS row stride for V chunk (64 ch x 32 l)

__device__ __forceinline__ void stage_kv_async(const bf16_t* __restrict__ Kr,
                                               const bf16_t* __restrict__ Vt,
                                               bf16_t* ksm, bf16_t* vsm,
                                               int b, int hkv, int kt, int tid) {
  {
    int row = tid >> 3, c8 = (tid & 7) << 3;   // 32 rows x 64 d, 16B per lane
    const bf16_t* g =
        Kr + (((size_t)(b * MAXKV + kt + row)) * HKV + hkv) * HD + c8;
    unsigned dst = (unsigned)(uintptr_t)(ksm + row * KPAD + c8);
    asm volatile("global_load_async_to_lds_b128 %0, %1, off"
                 :: "v"(dst), "v"(g) : "memory");
  }
  {
    int c = tid >> 2, l8 = (tid & 3) << 3;     // 64 ch x 32 l, 16B per lane
    const bf16_t* g =
        Vt + (((size_t)(b * HKV + hkv)) * HD + c) * MAXKV + kt + l8;
    unsigned dst = (unsigned)(uintptr_t)(vsm + c * VPAD + l8);
    asm volatile("global_load_async_to_lds_b128 %0, %1, off"
                 :: "v"(dst), "v"(g) : "memory");
  }
}

__global__ void __launch_bounds__(256)
flash_attn_wmma(const bf16_t* __restrict__ Qr, const bf16_t* __restrict__ Kr,
                const bf16_t* __restrict__ Vt, float* __restrict__ O,
                int koff) {
  __shared__ __align__(16) bf16_t Ksm[2][32 * KPAD];
  __shared__ __align__(16) bf16_t Vsm[2][64 * VPAD];
  __shared__ __align__(16) bf16_t psm[8 * 16 * 40];

  int tid = threadIdx.x;
  int wv = tid >> 5;
  int lane = tid & 31;
  int hf = lane >> 4, l16 = lane & 15;

  int qgrp = blockIdx.x & 7;          // 8 q-tile groups per (b,h)
  int h = (blockIdx.x >> 3) & 15;
  int b = blockIdx.x >> 7;
  int hkv = h >> 1;
  int q0 = qgrp * 128 + wv * 16;      // this wave's q-tile
  bf16_t* sm = psm + wv * (16 * 40);

  // preload Q fragments (d = 0..31, 32..63)
  const bf16_t* qrow =
      Qr + (((size_t)(b * TQ + q0 + l16)) * HQ + h) * HD + hf * 8;
  v16bf aq0 = ld_a_frag(qrow);
  v16bf aq1 = ld_a_frag(qrow + 32);

  float rmax[8], rsum[8];
  v8f oacc[4];
#pragma unroll
  for (int r = 0; r < 8; ++r) { rmax[r] = -1e30f; rsum[r] = 0.0f; }
#pragma unroll
  for (int g = 0; g < 4; ++g) oacc[g] = (v8f){};

  const float scale = 0.125f;                 // HD^-0.5
  int kend_wave = q0 + 16 + koff;             // this wave's causal bound
  int nchunks = (qgrp * 128 + 128 + koff) >> 5;   // block bound (multiple of 32)

  stage_kv_async(Kr, Vt, Ksm[0], Vsm[0], b, hkv, 0, tid);

  for (int ci = 0; ci < nchunks; ++ci) {
    int kt = ci << 5;
    int cur = ci & 1;
    asm volatile("s_wait_asynccnt 0x0" ::: "memory");
    __syncthreads();                          // chunk `cur` fully in LDS
    if (ci + 1 < nchunks)                     // overlap next chunk's DMA
      stage_kv_async(Kr, Vt, Ksm[cur ^ 1], Vsm[cur ^ 1], b, hkv, kt + 32, tid);

    if (kt < kend_wave) {                     // wave-uniform: EXEC stays all-1
      const bf16_t* kb0 = Ksm[cur] + l16 * KPAD + hf * 8;
      const bf16_t* kb1 = Ksm[cur] + (16 + l16) * KPAD + hf * 8;
      v8f s0 = {}, s1 = {};
      s0 = wmma_bf16(aq0, ld_b_frag(kb0), s0);
      s0 = wmma_bf16(aq1, ld_b_frag(kb0 + 32), s0);
      s1 = wmma_bf16(aq0, ld_b_frag(kb1), s1);
      s1 = wmma_bf16(aq1, ld_b_frag(kb1 + 32), s1);

#pragma unroll
      for (int r = 0; r < 8; ++r) {
        int qg = q0 + hf * 8 + r + koff;
        float v0 = (kt + l16      <= qg) ? s0[r] * scale : -1e30f;
        float v1 = (kt + 16 + l16 <= qg) ? s1[r] * scale : -1e30f;
        float mr = fmaxf(v0, v1);
        mr = fmaxf(mr, __shfl_xor(mr, 1));
        mr = fmaxf(mr, __shfl_xor(mr, 2));
        mr = fmaxf(mr, __shfl_xor(mr, 4));
        mr = fmaxf(mr, __shfl_xor(mr, 8));
        mr = fmaxf(mr, rmax[r]);
        float corr = __expf(rmax[r] - mr);
        rmax[r] = mr;
        float p0 = __expf(v0 - mr);
        float p1 = __expf(v1 - mr);
        float ps = p0 + p1;
        ps += __shfl_xor(ps, 1);
        ps += __shfl_xor(ps, 2);
        ps += __shfl_xor(ps, 4);
        ps += __shfl_xor(ps, 8);
        rsum[r] = rsum[r] * corr + ps;
        oacc[0][r] *= corr; oacc[1][r] *= corr;
        oacc[2][r] *= corr; oacc[3][r] *= corr;
        int row = hf * 8 + r;
        sm[row * 40 + l16]      = (bf16_t)p0;
        sm[row * 40 + 16 + l16] = (bf16_t)p1;
      }

      // reload P as A fragment (same-wave LDS ops are in order)
      v16bf ap = ld_a_frag(sm + l16 * 40 + hf * 8);

      // O += P (16x32) . V (32x64)
#pragma unroll
      for (int g = 0; g < 4; ++g)
        oacc[g] = wmma_bf16(
            ap, ld_b_frag(Vsm[cur] + (g * 16 + l16) * VPAD + hf * 8), oacc[g]);
    }
  }

  // ---- normalize and store ----
#pragma unroll
  for (int g = 0; g < 4; ++g) {
#pragma unroll
    for (int r = 0; r < 8; ++r) {
      int q = q0 + hf * 8 + r;
      O[(((size_t)(b * TQ + q)) * HQ + h) * HD + g * 16 + l16] =
          oacc[g][r] / rsum[r];
    }
  }
}

// ---------- acc += rmsnorm(o, ln_w[i]) * lw[i] ; one block per row ----------

__global__ void __launch_bounds__(256)
accum_rmsnorm_kernel(const float* __restrict__ O, const float* __restrict__ ln_w,
                     const float* __restrict__ lam, float* __restrict__ acc,
                     int layer) {
  __shared__ float red[256];
  int row = blockIdx.x;
  int tid = threadIdx.x;
  float ss = 0.0f;
  for (int d = tid; d < DQ; d += 256) {
    float v = O[(size_t)row * DQ + d];
    ss += v * v;
  }
  red[tid] = ss;
  __syncthreads();
  for (int s = 128; s > 0; s >>= 1) {
    if (tid < s) red[tid] += red[tid + s];
    __syncthreads();
  }
  float rms = rsqrtf(red[0] / (float)DQ + EPSQ);
  float s0 = 1.0f / (1.0f + __expf(-lam[0]));
  float s1 = 1.0f / (1.0f + __expf(-lam[1]));
  float s2 = 1.0f / (1.0f + __expf(-lam[2]));
  float mean = (s0 + s1 + s2) * (1.0f / 3.0f);
  float var = ((s0 - mean) * (s0 - mean) + (s1 - mean) * (s1 - mean) +
               (s2 - mean) * (s2 - mean)) * (1.0f / 3.0f);
  float sl = (layer == 0) ? s0 : (layer == 1 ? s1 : s2);
  float lw = (sl - mean) * rsqrtf(var + EPSQ);
  for (int d = tid; d < DQ; d += 256) {
    acc[(size_t)row * DQ + d] +=
        O[(size_t)row * DQ + d] * rms * ln_w[layer * DQ + d] * lw;
  }
}

// ---------- y_bf16 = rmsnorm(acc + alpha*x, final_ln_w) ----------

__global__ void __launch_bounds__(256)
final_norm_kernel(const float* __restrict__ acc, const float* __restrict__ x,
                  const float* __restrict__ alpha, const float* __restrict__ fw,
                  bf16_t* __restrict__ y) {
  __shared__ float red[256];
  int row = blockIdx.x;
  int tid = threadIdx.x;
  float a = alpha[0];
  float ss = 0.0f;
  for (int d = tid; d < DQ; d += 256) {
    float v = acc[(size_t)row * DQ + d] + a * x[(size_t)row * DQ + d];
    ss += v * v;
  }
  red[tid] = ss;
  __syncthreads();
  for (int s = 128; s > 0; s >>= 1) {
    if (tid < s) red[tid] += red[tid + s];
    __syncthreads();
  }
  float rms = rsqrtf(red[0] / (float)DQ + EPSQ);
  for (int d = tid; d < DQ; d += 256) {
    float v = acc[(size_t)row * DQ + d] + a * x[(size_t)row * DQ + d];
    y[(size_t)row * DQ + d] = (bf16_t)(v * rms * fw[d]);
  }
}

// ---------------------------- host launcher ----------------------------

extern "C" void kernel_launch(void* const* d_in, const int* in_sizes, int n_in,
                              void* d_out, int out_size, void* d_ws, size_t ws_size,
                              hipStream_t stream) {
  const float* x     = (const float*)d_in[0];
  const float* cosT  = (const float*)d_in[1];
  const float* sinT  = (const float*)d_in[2];
  const float* q_w   = (const float*)d_in[3];
  const float* k_w   = (const float*)d_in[4];
  const float* v_w   = (const float*)d_in[5];
  const float* ln_w  = (const float*)d_in[6];
  const float* lam   = (const float*)d_in[7];
  const float* out_w = (const float*)d_in[8];
  const float* fln   = (const float*)d_in[9];
  const float* alpha = (const float*)d_in[10];
  float* out = (float*)d_out;

  char* ws = (char*)d_ws;
  size_t cur = 0;
  auto take = [&](size_t bytes) -> char* {
    char* p = ws + cur;
    cur += (bytes + 255) & ~(size_t)255;
    return p;
  };

  const int M = BQ * TQ;             // 2048
  bf16_t* xb   = (bf16_t*)take((size_t)M * DQ * 2);
  bf16_t* qwb  = (bf16_t*)take((size_t)3 * DQ * DQ * 2);
  bf16_t* kwb  = (bf16_t*)take((size_t)3 * 512 * DQ * 2);
  bf16_t* vwb  = (bf16_t*)take((size_t)3 * 512 * DQ * 2);
  bf16_t* owb  = (bf16_t*)take((size_t)DQ * DQ * 2);
  float*  Qf   = (float*) take((size_t)M * DQ * 4);
  bf16_t* Qrb  = (bf16_t*)take((size_t)M * DQ * 2);
  float*  Kc   = (float*) take((size_t)BQ * MAXKV * 512 * 4);
  bf16_t* Krb  = (bf16_t*)take((size_t)BQ * MAXKV * 512 * 2);
  float*  Vtmp = (float*) take((size_t)M * 512 * 4);
  bf16_t* Vtb  = (bf16_t*)take((size_t)BQ * HKV * HD * MAXKV * 2);
  float*  Of   = (float*) take((size_t)M * DQ * 4);
  float*  Acc  = (float*) take((size_t)M * DQ * 4);
  bf16_t* Yb   = (bf16_t*)take((size_t)M * DQ * 2);

  auto cvt = [&](const float* s, bf16_t* d, int n) {
    f32_to_bf16_kernel<<<(n + 255) / 256, 256, 0, stream>>>(s, d, n);
  };
  cvt(x, xb, M * DQ);
  cvt(q_w, qwb, 3 * DQ * DQ);
  cvt(k_w, kwb, 3 * 512 * DQ);
  cvt(v_w, vwb, 3 * 512 * DQ);
  cvt(out_w, owb, DQ * DQ);
  zero_f32_kernel<<<(M * DQ + 255) / 256, 256, 0, stream>>>(Acc, M * DQ);

  for (int i = 0; i < 3; ++i) {
    int L = (i + 1) * TQ;
    int koff = i * TQ;
    // Q = x @ q_w[i]^T  -> (B*T, 1024)   tiles = 32x32 -> 128 blocks
    wmma_gemm_bf16<<<128, 256, 0, stream>>>(
        xb, qwb + (size_t)i * DQ * DQ, Qf, M, DQ, DQ, TQ, TQ, 0);
    // K -> KV cache slot rows [i*T, (i+1)*T)   tiles = 32x16 -> 64 blocks
    wmma_gemm_bf16<<<64, 256, 0, stream>>>(
        xb, kwb + (size_t)i * 512 * DQ, Kc, M, 512, DQ, TQ, MAXKV, koff);
    // V -> temp
    wmma_gemm_bf16<<<64, 256, 0, stream>>>(
        xb, vwb + (size_t)i * 512 * DQ, Vtmp, M, 512, DQ, TQ, TQ, 0);

    int nq = BQ * TQ * HQ * 32;
    rope_q_kernel<<<(nq + 255) / 256, 256, 0, stream>>>(Qf, cosT, sinT, Qrb, nq);
    int nk = BQ * L * HKV * 32;
    rope_k_kernel<<<(nk + 255) / 256, 256, 0, stream>>>(Kc, cosT, sinT, Krb, L, nk);
    int nv = BQ * TQ * HKV * HD;
    v_transpose_kernel<<<(nv + 255) / 256, 256, 0, stream>>>(Vtmp, Vtb, koff, nv);

    // 256 blocks: (b,h,qgrp) ; 8 waves share K/V chunks via async LDS staging
    flash_attn_wmma<<<BQ * HQ * 8, 256, 0, stream>>>(Qrb, Krb, Vtb, Of, koff);

    accum_rmsnorm_kernel<<<M, 256, 0, stream>>>(Of, ln_w, lam, Acc, i);
  }

  final_norm_kernel<<<M, 256, 0, stream>>>(Acc, x, alpha, fln, Yb);
  // out = y @ out_w^T
  wmma_gemm_bf16<<<128, 256, 0, stream>>>(Yb, owb, out, M, DQ, DQ, TQ, TQ, 0);
}
